// BertForSemanticEmbedding_16973710754315
// MI455X (gfx1250) — compile-verified, matrix operand order checked by reading, not run on hardware
//
#include <hip/hip_runtime.h>
#include <hip/hip_bf16.h>

#define TOKEN_DIM 32
#define NQ 16
#define LQ 128
#define ND 128
#define LD 128

typedef __attribute__((ext_vector_type(16))) _Float16 v16h;
typedef __attribute__((ext_vector_type(8)))  float    v8f;

// One workgroup per (q, n) pair. 8 waves; wave w owns query-token block
// i0 = 16*w. Doc-side WMMA B fragments are shared across waves -> staged in
// LDS once (f32 -> f16 conversion done once per workgroup, not per wave).
__global__ __launch_bounds__(256)
void coil_maxsim_wmma(const float* __restrict__ doc_reps,   // [ND, LD, 32]
                      const float* __restrict__ qry_reps,   // [NQ, LQ, 32]
                      const float* __restrict__ qmask,      // [NQ, LQ]
                      const int*   __restrict__ doc_ids,    // [ND, LD]
                      const int*   __restrict__ qry_ids,    // [NQ, LQ]
                      float*       __restrict__ out)        // [NQ, ND]
{
    const int q    = blockIdx.x >> 7;        // blockIdx = q*ND + n
    const int n    = blockIdx.x & (ND - 1);
    const int wave = threadIdx.x >> 5;       // 0..7
    const int lane = threadIdx.x & 31;
    const int half = lane >> 4;              // selects K-half of the fragment
    const int l15  = lane & 15;              // tile row (load) / N column (C)

    __shared__ v16h  bfrag[8][32];           // 8 j-tiles x 32 lanes x 32B = 8KB
    __shared__ float partial[8];

    // ---- Stage doc-side B fragments in LDS (wave w converts j-tile w) ----
    // B is K x N = 32 x 16 f16: lane L<16 holds column N=L with K=0..7,16..23;
    // lanes >=16 hold K=8..15,24..31. B[k][j] = doc_reps[n, j, k], so each
    // lane gathers two contiguous 8-float chunks of doc token row j.
    {
        const float* drow =
            doc_reps + (size_t)(n * LD + wave * 16 + l15) * TOKEN_DIM + half * 8;
        v16h b;
#pragma unroll
        for (int t = 0; t < 8; ++t) b[t]     = (_Float16)drow[t];
#pragma unroll
        for (int t = 0; t < 8; ++t) b[8 + t] = (_Float16)drow[16 + t];
        bfrag[wave][lane] = b;               // ds_store_b128 x2
    }

    // ---- A fragment (query side) for this wave's i-block ----
    const int i0 = wave * 16;
    v16h a;
    {
        const float* qrow =
            qry_reps + (size_t)(q * LQ + i0 + l15) * TOKEN_DIM + half * 8;
#pragma unroll
        for (int t = 0; t < 8; ++t) a[t]     = (_Float16)qrow[t];
#pragma unroll
        for (int t = 0; t < 8; ++t) a[8 + t] = (_Float16)qrow[16 + t];
    }

    // ---- Per-lane query ids + CLS-masked weights for its 8 accumulator rows.
    // C layout: VGPR r holds M = r + 8*half (lane group), N = l15.
    int   qid[8];
    float wgt[8];
#pragma unroll
    for (int r = 0; r < 8; ++r) {
        const int i = i0 + r + half * 8;
        qid[r] = qry_ids[q * LQ + i];
        wgt[r] = (i == 0) ? 0.0f : qmask[q * LQ + i];
    }

    __syncthreads();

    float m[8];
#pragma unroll
    for (int r = 0; r < 8; ++r) m[r] = -3.402823466e38f;

    // ---- Fused GEMM + exact-match mask + running MaxSim over j-tiles ----
#pragma unroll
    for (int jt = 0; jt < 8; ++jt) {
        v16h b = bfrag[jt][lane];            // ds_load_b128 x2
        v8f  c = {};
        // D = A x B + 0 : v_wmma_f32_16x16x32_f16
        c = __builtin_amdgcn_wmma_f32_16x16x32_f16(
                /*neg_a=*/false, a, /*neg_b=*/false, b,
                /*c_mod=*/(short)0, c, /*reuse_a=*/false, /*reuse_b=*/false);

        const int j   = jt * 16 + l15;
        const int did = doc_ids[n * LD + j];
#pragma unroll
        for (int r = 0; r < 8; ++r) {
            const float v = (qid[r] == did) ? c[r] : 0.0f;  // where(match, s, 0)
            m[r] = fmaxf(m[r], v);                          // max over j (partial)
        }
    }

    // ---- Max-reduce across the 16 lanes that hold disjoint j columns ----
    // (xor distances <16 stay within each half-wave, whose rows differ.)
#pragma unroll
    for (int r = 0; r < 8; ++r) {
        float v = m[r];
        v = fmaxf(v, __shfl_xor(v, 1, 32));
        v = fmaxf(v, __shfl_xor(v, 2, 32));
        v = fmaxf(v, __shfl_xor(v, 4, 32));
        v = fmaxf(v, __shfl_xor(v, 8, 32));
        m[r] = v;
    }

    // ---- Weighted sum over query tokens (skip i==0 via wgt) ----
    float s = 0.0f;
#pragma unroll
    for (int r = 0; r < 8; ++r) s += m[r] * wgt[r];
    s += __shfl_xor(s, 16, 32);   // combine i0..i0+7 with i0+8..i0+15

    if (lane == 0) partial[wave] = s;
    __syncthreads();
    if (threadIdx.x == 0) {
        float tot = 0.0f;
#pragma unroll
        for (int w = 0; w < 8; ++w) tot += partial[w];
        out[q * ND + n] = tot;    // full overwrite: deterministic, no atomics
    }
}

extern "C" void kernel_launch(void* const* d_in, const int* in_sizes, int n_in,
                              void* d_out, int out_size, void* d_ws, size_t ws_size,
                              hipStream_t stream) {
    (void)in_sizes; (void)n_in; (void)out_size; (void)d_ws; (void)ws_size;
    const float* doc_reps = (const float*)d_in[0];
    const float* qry_reps = (const float*)d_in[1];
    const float* qmask    = (const float*)d_in[2];
    const int*   doc_ids  = (const int*)d_in[3];
    const int*   qry_ids  = (const int*)d_in[4];
    float*       out      = (float*)d_out;

    dim3 grid(NQ * ND);   // 2048 workgroups, one per (q, n)
    dim3 block(256);      // 8 wave32s
    hipLaunchKernelGGL(coil_maxsim_wmma, grid, block, 0, stream,
                       doc_reps, qry_reps, qmask, doc_ids, qry_ids, out);
}
